// MultiheadSelfAttention_47038481826244
// MI455X (gfx1250) — compile-verified, matrix operand order checked by reading, not run on hardware
//
#include <hip/hip_runtime.h>
#include <hip/hip_bf16.h>

typedef __attribute__((ext_vector_type(8)))  __bf16       v8bf;
typedef __attribute__((ext_vector_type(16))) __bf16       v16bf;
typedef __attribute__((ext_vector_type(8)))  float        v8f;
typedef __attribute__((ext_vector_type(4)))  unsigned int v4u;
typedef __attribute__((ext_vector_type(8)))  int          v8i;
typedef __attribute__((ext_vector_type(4)))  int          v4i;

__device__ inline v16bf mk16(v8bf lo, v8bf hi) {
  return __builtin_shufflevector(lo, hi, 0,1,2,3,4,5,6,7,8,9,10,11,12,13,14,15);
}

__device__ inline v8f wmma_bf16(v16bf a, v16bf b, v8f c) {
  // D(16x16,f32) = A(16x32,bf16) * B(32x16,bf16) + C
  return __builtin_amdgcn_wmma_f32_16x16x32_bf16(false, a, false, b,
                                                 (short)0, c, false, false);
}

// ---------------------------------------------------------------------------
// Tensor Data Mover: async 2-D tile load (bf16) global -> LDS.
// D# per CDNA5 ISA ch.8: group0 = {flags, lds_addr, global_addr, type=2},
// group1 = {data_size=2B, tensor_dim0/1, tile_dim0/1, dim0_stride}.
// Tracked by TENSORcnt; LDS receives tile_d1 rows x tile_d0 elems, row-major.
// ---------------------------------------------------------------------------
__device__ inline void tdm_load_2d(unsigned lds_off, const void* gptr,
                                   unsigned tensor_d0, unsigned tensor_d1,
                                   unsigned tile_d0, unsigned tile_d1,
                                   unsigned stride0) {
  unsigned long long ga = (unsigned long long)(uintptr_t)gptr;
  v4u g0;
  g0[0] = 1u;                                        // count=1 (valid user D#)
  g0[1] = lds_off;                                   // lds_addr (bytes)
  g0[2] = (unsigned)(ga & 0xffffffffu);              // global_addr[31:0]
  g0[3] = (unsigned)((ga >> 32) & 0x01ffffffu)       // global_addr[56:32]
        | (2u << 30);                                // type = 2 ("image")
  v8i g1;
  g1[0] = (int)(1u << 16);                           // data_size=1 -> 2 bytes
  g1[1] = (int)((tensor_d0 & 0xffffu) << 16);        // tensor_dim0[15:0]
  g1[2] = (int)((tensor_d0 >> 16) | ((tensor_d1 & 0xffffu) << 16));
  g1[3] = (int)((tensor_d1 >> 16) | ((tile_d0 & 0xffffu) << 16));
  g1[4] = (int)(tile_d1 & 0xffffu);                  // tile_dim1; tile_dim2=0
  g1[5] = (int)stride0;                              // dim0_stride[31:0]
  g1[6] = 0;                                         // stride hi / dim1_stride
  g1[7] = 0;
  v4i z = {0, 0, 0, 0};                              // groups 2/3: 2-D tensor
#if __clang_major__ >= 23
  v8i z8 = {0, 0, 0, 0, 0, 0, 0, 0};
  __builtin_amdgcn_tensor_load_to_lds(g0, g1, z, z, z8, 0);
#else
  __builtin_amdgcn_tensor_load_to_lds(g0, g1, z, z, 0);
#endif
}

__device__ inline unsigned lds_off_of(const void* p) {
  return (unsigned)(uintptr_t)p;                     // low 32 bits = LDS offset
}

// ---------------------------------------------------------------------------
// fp32 -> bf16 elementwise convert
// ---------------------------------------------------------------------------
__global__ void f32_to_bf16_kernel(const float* __restrict__ in,
                                   __bf16* __restrict__ out, int n) {
  int i = blockIdx.x * blockDim.x + threadIdx.x;
  if (i < n) out[i] = (__bf16)in[i];
}

// ---------------------------------------------------------------------------
// GEMM: Y[M,1024] = A[M,1024] @ W[1024,1024]^T.
// Block = 256 threads (8 waves) computes a 128x128 tile. K-chunks of 64 are
// TDM-DMA'd into double-buffered LDS (A 16KB + B 16KB per buffer = 64KB);
// wave0 produces, s_wait_tensorcnt + barrier publishes, all waves consume
// via ds_load_b128. Each wave: 32(M)x64(N), 16 WMMA per stage, 16 stages.
// mode 0: bf16 out [b,h,s,d] (Q/K)   mode 1: bf16 out [b,h,d,s] (V^T)
// mode 2: fp32 out [M,1024]  (final projection)
// ---------------------------------------------------------------------------
__global__ __launch_bounds__(256)
void gemm_bf16_wmma(const __bf16* __restrict__ A,
                    const __bf16* __restrict__ W,
                    void* __restrict__ out, int M, int mode) {
  __shared__ __bf16 lA[2][128 * 64];   // [row m][k] row-major, 16KB per buffer
  __shared__ __bf16 lB[2][128 * 64];   // [row n][k] row-major (rows of W)

  const int KD = 1024;
  int lane = threadIdx.x & 31;
  int w    = threadIdx.x >> 5;
  int col  = lane & 15;
  int hi   = lane >> 4;
  int kb   = hi * 8;                   // per-lane K sub-base (ISA A/B layout)

  int mtile = blockIdx.x >> 3;         // 1024/128 = 8 N-tiles
  int ntile = blockIdx.x & 7;
  int m0 = mtile * 128, n0 = ntile * 128;
  int mw = (w & 3) * 32;               // wave strip: 4 waves over M
  int nw = (w >> 2) * 64;              // 2 waves over N
  bool producer = (w == 0);

  if (producer) {
    tdm_load_2d(lds_off_of(&lA[0][0]), A + (size_t)m0 * KD, KD, (unsigned)M, 64, 128, KD);
    tdm_load_2d(lds_off_of(&lB[0][0]), W + (size_t)n0 * KD, KD, 1024u,       64, 128, KD);
  }

  v8f acc[2][4] = {};
  for (int ks = 0; ks < 16; ++ks) {    // 1024 / 64-wide K stages
    int cur = ks & 1;
    if (producer) __builtin_amdgcn_s_wait_tensorcnt(0);
    __syncthreads();                                  // publish buf[cur]
    if (producer && ks + 1 < 16) {                    // prefetch buf[1-cur]
      int k = (ks + 1) * 64;
      tdm_load_2d(lds_off_of(&lA[1 - cur][0]), A + (size_t)m0 * KD + k, KD, (unsigned)M, 64, 128, KD);
      tdm_load_2d(lds_off_of(&lB[1 - cur][0]), W + (size_t)n0 * KD + k, KD, 1024u,       64, 128, KD);
    }
    const __bf16* la = &lA[cur][0];
    const __bf16* lb = &lB[cur][0];
#pragma unroll
    for (int kk = 0; kk < 64; kk += 32) {
      v16bf aop[2];
#pragma unroll
      for (int mi = 0; mi < 2; ++mi) {
        int row = mw + mi * 16 + col;
        aop[mi] = mk16(*(const v8bf*)(la + row * 64 + kk + kb),
                       *(const v8bf*)(la + row * 64 + kk + 16 + kb));
      }
#pragma unroll
      for (int j = 0; j < 4; ++j) {
        int row = nw + j * 16 + col;
        v16bf b = mk16(*(const v8bf*)(lb + row * 64 + kk + kb),
                       *(const v8bf*)(lb + row * 64 + kk + 16 + kb));
#pragma unroll
        for (int mi = 0; mi < 2; ++mi)
          acc[mi][j] = wmma_bf16(aop[mi], b, acc[mi][j]);
      }
    }
    __syncthreads();                                  // release buf[cur]
  }

  // C/D layout: VGPR r, lanes 0-15 -> M=r, lanes 16-31 -> M=r+8; N = lane&15
#pragma unroll
  for (int mi = 0; mi < 2; ++mi)
#pragma unroll
    for (int r = 0; r < 8; ++r) {
      int m = m0 + mw + mi * 16 + r + 8 * hi;        // global row (= b*2048+s)
      int b = m >> 11, s = m & 2047;
#pragma unroll
      for (int j = 0; j < 4; ++j) {
        int n = n0 + nw + j * 16 + col;              // global col (= h*64+d)
        float v = acc[mi][j][r];
        if (mode == 0) {
          int h = n >> 6, d = n & 63;
          ((__bf16*)out)[(((size_t)(b * 16 + h) * 2048) + s) * 64 + d] = (__bf16)v;
        } else if (mode == 1) {
          int h = n >> 6, d = n & 63;
          ((__bf16*)out)[(((size_t)(b * 16 + h) * 64) + d) * 2048 + s] = (__bf16)v;
        } else {
          ((float*)out)[(size_t)m * 1024 + n] = v;
        }
      }
    }
}

// ---------------------------------------------------------------------------
// Flash attention. Block = 8 waves, ALL on the same (b,h): wave w owns query
// block sblk = qg*8+w. 64-key K/V chunks are TDM-DMA'd into double-buffered
// LDS once per block (K: [64 t][64 d], V^T: [64 d][64 t], 8KB each) and
// consumed by all 8 waves. Online softmax state in VGPRs; P re-laid-out
// (D-layout -> A-layout) through a per-wave LDS slot.
// ---------------------------------------------------------------------------
__global__ __launch_bounds__(256)
void flash_attn_wmma(const __bf16* __restrict__ Q,
                     const __bf16* __restrict__ K,
                     const __bf16* __restrict__ Vt,
                     __bf16* __restrict__ O) {
  __shared__ __bf16 lK[2][64 * 64];    // [t][d] rows, 8KB per buffer
  __shared__ __bf16 lV[2][64 * 64];    // [d][t] rows, 8KB per buffer
  __shared__ __bf16 ldsP[8][16 * 32];  // per-wave P staging tile

  int lane = threadIdx.x & 31;
  int slot = threadIdx.x >> 5;
  int bh   = blockIdx.x >> 4;          // 32 batch-heads
  int qg   = blockIdx.x & 15;          // 16 query-groups of 8 blocks
  int sblk = qg * 8 + slot;
  int col  = lane & 15;
  int hi   = lane >> 4;
  int kb   = hi * 8;
  bool producer = (slot == 0);

  const __bf16* Qb = Q  + (size_t)bh * 2048 * 64;
  const __bf16* Kb = K  + (size_t)bh * 2048 * 64;
  const __bf16* Vb = Vt + (size_t)bh * 64 * 2048;

  if (producer) {
    tdm_load_2d(lds_off_of(&lK[0][0]), Kb, 64,   2048, 64, 64, 64);
    tdm_load_2d(lds_off_of(&lV[0][0]), Vb, 2048, 64,   64, 64, 2048);
  }

  // Q A-operand: lane holds query row (sblk*16+col), d split per ISA layout
  const __bf16* qrow = Qb + (size_t)(sblk * 16 + col) * 64;
  v16bf qa0 = mk16(*(const v8bf*)(qrow + kb),      *(const v8bf*)(qrow + 16 + kb));
  v16bf qa1 = mk16(*(const v8bf*)(qrow + 32 + kb), *(const v8bf*)(qrow + 48 + kb));

  float rmax[8], rsum[8];
#pragma unroll
  for (int r = 0; r < 8; ++r) { rmax[r] = -1e30f; rsum[r] = 0.f; }
  v8f o[4] = {};

  for (int ks = 0; ks < 32; ++ks) {    // 2048 keys / 64 per stage
    int cur = ks & 1;
    if (producer) __builtin_amdgcn_s_wait_tensorcnt(0);
    __syncthreads();                   // publish K/V buf[cur]
    if (producer && ks + 1 < 32) {
      int t = (ks + 1) * 64;
      tdm_load_2d(lds_off_of(&lK[1 - cur][0]), Kb + (size_t)t * 64, 64,   2048, 64, 64, 64);
      tdm_load_2d(lds_off_of(&lV[1 - cur][0]), Vb + t,              2048, 64,   64, 64, 2048);
    }
    const __bf16* kt = &lK[cur][0];
    const __bf16* vt = &lV[cur][0];

#pragma unroll
    for (int c = 0; c < 2; ++c) {      // two 32-key sub-chunks per stage
      int tc = c * 32;
      // ---- scores: two 16x16 tiles over this 32-key sub-chunk -------------
      v8f c0 = {}, c1 = {};
      {
        int t = tc + col;                          // key tile 0
        v16bf b0 = mk16(*(const v8bf*)(kt + t * 64 + kb),      *(const v8bf*)(kt + t * 64 + 16 + kb));
        v16bf b1 = mk16(*(const v8bf*)(kt + t * 64 + 32 + kb), *(const v8bf*)(kt + t * 64 + 48 + kb));
        c0 = wmma_bf16(qa0, b0, c0);
        c0 = wmma_bf16(qa1, b1, c0);
      }
      {
        int t = tc + 16 + col;                     // key tile 1
        v16bf b0 = mk16(*(const v8bf*)(kt + t * 64 + kb),      *(const v8bf*)(kt + t * 64 + 16 + kb));
        v16bf b1 = mk16(*(const v8bf*)(kt + t * 64 + 32 + kb), *(const v8bf*)(kt + t * 64 + 48 + kb));
        c1 = wmma_bf16(qa0, b0, c1);
        c1 = wmma_bf16(qa1, b1, c1);
      }

      // ---- online softmax (row reductions across 16-lane groups) ----------
      float alpha[8];
#pragma unroll
      for (int r = 0; r < 8; ++r) {
        c0[r] *= 0.125f;                           // 1/sqrt(64)
        c1[r] *= 0.125f;
        float v = fmaxf(c0[r], c1[r]);
        v = fmaxf(v, __shfl_xor(v, 1, 32));
        v = fmaxf(v, __shfl_xor(v, 2, 32));
        v = fmaxf(v, __shfl_xor(v, 4, 32));
        v = fmaxf(v, __shfl_xor(v, 8, 32));
        float mnew = fmaxf(rmax[r], v);
        alpha[r] = __expf(rmax[r] - mnew);
        rmax[r]  = mnew;
        float p0 = __expf(c0[r] - mnew);
        float p1 = __expf(c1[r] - mnew);
        c0[r] = p0; c1[r] = p1;
        float s = p0 + p1;
        s += __shfl_xor(s, 1, 32);
        s += __shfl_xor(s, 2, 32);
        s += __shfl_xor(s, 4, 32);
        s += __shfl_xor(s, 8, 32);
        rsum[r] = rsum[r] * alpha[r] + s;
      }
#pragma unroll
      for (int j = 0; j < 4; ++j)
#pragma unroll
        for (int r = 0; r < 8; ++r) o[j][r] *= alpha[r];

      // ---- D-layout P -> per-wave LDS slot -> A-layout P -------------------
#pragma unroll
      for (int r = 0; r < 8; ++r) {
        int m = r + 8 * hi;
        ldsP[slot][m * 32 + col]      = (__bf16)c0[r];
        ldsP[slot][m * 32 + 16 + col] = (__bf16)c1[r];
      }
      // per-wave slot; DS ops are in-order within a wave -> no barrier needed
      v16bf pa = mk16(*(const v8bf*)(&ldsP[slot][col * 32 + kb]),
                      *(const v8bf*)(&ldsP[slot][col * 32 + 16 + kb]));

      // ---- O += P @ V  (V^T rows are d, contiguous in t) -------------------
#pragma unroll
      for (int j = 0; j < 4; ++j) {
        int n = j * 16 + col;
        v16bf vb = mk16(*(const v8bf*)(vt + n * 64 + tc + kb),
                        *(const v8bf*)(vt + n * 64 + tc + 16 + kb));
        o[j] = wmma_bf16(pa, vb, o[j]);
      }
    }
    __syncthreads();                   // release K/V buf[cur]
  }

  // ---- normalize and store as [b, s, h*64 + d] (bf16) ----------------------
  int b = bh >> 4, h = bh & 15;
#pragma unroll
  for (int r = 0; r < 8; ++r) {
    int s = sblk * 16 + r + 8 * hi;
    float inv = 1.0f / rsum[r];
#pragma unroll
    for (int j = 0; j < 4; ++j) {
      int c = h * 64 + j * 16 + col;
      O[(size_t)(b * 2048 + s) * 1024 + c] = (__bf16)(o[j][r] * inv);
    }
  }
}

// ---------------------------------------------------------------------------
extern "C" void kernel_launch(void* const* d_in, const int* in_sizes, int n_in,
                              void* d_out, int out_size, void* d_ws, size_t ws_size,
                              hipStream_t stream) {
  const float* x  = (const float*)d_in[0];
  const float* Wq = (const float*)d_in[1];
  const float* Wk = (const float*)d_in[2];
  const float* Wv = (const float*)d_in[3];
  const float* Wo = (const float*)d_in[4];

  const size_t MiB = 1ull << 20;
  if (ws_size < 48 * MiB) return;            // need 48 MiB of scratch
  char* ws = (char*)d_ws;
  __bf16* xb  = (__bf16*)(ws + 0 * MiB);     // [4096,1024]   8 MiB
  __bf16* wqb = (__bf16*)(ws + 8 * MiB);     // [1024,1024]   2 MiB
  __bf16* wkb = (__bf16*)(ws + 10 * MiB);
  __bf16* wvb = (__bf16*)(ws + 12 * MiB);
  __bf16* wob = (__bf16*)(ws + 14 * MiB);
  __bf16* Qb  = (__bf16*)(ws + 16 * MiB);    // [32,2048,64]  8 MiB
  __bf16* Kb  = (__bf16*)(ws + 24 * MiB);    // [32,2048,64]  8 MiB
  __bf16* Vt  = (__bf16*)(ws + 32 * MiB);    // [32,64,2048]  8 MiB
  __bf16* Ob  = (__bf16*)(ws + 40 * MiB);    // [4096,1024]   8 MiB

  const int nx = 4096 * 1024, nw = 1024 * 1024;
  f32_to_bf16_kernel<<<nx / 256, 256, 0, stream>>>(x,  xb,  nx);
  f32_to_bf16_kernel<<<nw / 256, 256, 0, stream>>>(Wq, wqb, nw);
  f32_to_bf16_kernel<<<nw / 256, 256, 0, stream>>>(Wk, wkb, nw);
  f32_to_bf16_kernel<<<nw / 256, 256, 0, stream>>>(Wv, wvb, nw);
  f32_to_bf16_kernel<<<nw / 256, 256, 0, stream>>>(Wo, wob, nw);

  // projections: (4096/128) x (1024/128) = 256 blocks of 128x128 tiles
  const int gblocks = (4096 / 128) * (1024 / 128);
  gemm_bf16_wmma<<<gblocks, 256, 0, stream>>>(xb, wqb, Qb, 4096, 0);
  gemm_bf16_wmma<<<gblocks, 256, 0, stream>>>(xb, wkb, Kb, 4096, 0);
  gemm_bf16_wmma<<<gblocks, 256, 0, stream>>>(xb, wvb, Vt, 4096, 1);

  // attention: 32 bh x 16 query-groups = 512 blocks (8 waves, same bh)
  flash_attn_wmma<<<512, 256, 0, stream>>>(Qb, Kb, Vt, Ob);

  // output projection (fp32 result)
  gemm_bf16_wmma<<<gblocks, 256, 0, stream>>>(Ob, wob, d_out, 4096, 2);
}
